// ScaledDotProductAttention_58609123721763
// MI455X (gfx1250) — matrix operand order repaired and checked by reading, not compiled, not run on hardware
//
#include <hip/hip_runtime.h>
#include <hip/hip_fp16.h>

typedef __attribute__((ext_vector_type(16))) _Float16 v16h;
typedef __attribute__((ext_vector_type(2)))  __fp16   v2hf;   // cvt_pkrtz result type
typedef __attribute__((ext_vector_type(8)))  float    v8f;
typedef __attribute__((ext_vector_type(4)))  unsigned int v4u;
typedef __attribute__((ext_vector_type(2)))  unsigned int v2u;
typedef __attribute__((ext_vector_type(4)))  int      v4i;
typedef __attribute__((ext_vector_type(8)))  int      v8i;

#define B_      8
#define LQ      2048
#define LK      2048
#define DH      128
#define QTILE   16
#define SSTRIDE (LK + 4)              // fp32 strip stride: float4-aligned, 4-bank skew
#define PSTRIDE (LK + 8)              // f16 prob strip stride (halves): 16B-aligned, 4-bank skew
#define KSTR    (DH + 4)              // TDM-padded K tile row stride (floats)
#define VSTR    20                    // TDM-padded V tile row stride (floats)
#define SCALE   0.08838834764831845f  // 1/sqrt(128)
#define NEGINF  (-1.0e9f)

// ---- shared-memory arena (267,648 B total) ----
#define OFF_REGB   131328             // after fp32 strip (16*2052*4)
#define KBUF_SZ    8448               // 16 rows * 132 floats (TDM pad 4 DW / 128 DW)
#define OFF_VST    (OFF_REGB + 65792) // after f16 P-strip (16*2056*2)
#define VBUF_SZ    2560               // 32 rows * 20 floats (TDM pad 4 DW / 16 DW)
#define OFF_RED    (OFF_REGB + 135168)
#define OFF_RMAX   (OFF_RED + 1024)
#define OFF_RRCP   (OFF_RMAX + 64)
#define ARENA_SZ   (OFF_RRCP + 64)

union HV { v16h v; v2hf h[8]; v4u q[2]; };
union PK { v2hf h[2]; v2u u; };

__device__ __forceinline__ unsigned lds_off(const void* p) {
    return (unsigned)(unsigned long long)p;   // low 32 bits of flat addr = LDS offset
}

// 2D-tile TDM load (fp32) global -> LDS with optional per-row LDS padding.
__device__ __forceinline__ void tdm_load_2d(unsigned lds_addr, const void* gptr,
                                            unsigned tensor_d0, unsigned tensor_d1,
                                            unsigned tile_d0, unsigned tile_d1,
                                            unsigned stride0_elems,
                                            unsigned pad_interval_code,
                                            unsigned pad_amount_code)
{
    unsigned long long ga = (unsigned long long)gptr;
    v4u g0;
    g0[0] = 1u;                                                // count=1, user desc
    g0[1] = lds_addr;
    g0[2] = (unsigned)ga;                                      // global_addr[31:0]
    g0[3] = (unsigned)((ga >> 32) & 0x01FFFFFFu) | (2u << 30); // ga[56:32] | type=2
    v8i g1;
    g1[0] = (int)((2u << 16) | (1u << 20) |                    // data_size=4B, pad_enable
                  (pad_interval_code << 22) | (pad_amount_code << 25));
    g1[1] = (int)(tensor_d0 << 16);
    g1[2] = (int)(tensor_d1 << 16);
    g1[3] = (int)(tile_d0 << 16);
    g1[4] = (int)tile_d1;
    g1[5] = (int)stride0_elems;
    g1[6] = 0;
    g1[7] = 0;
    v4i z4 = {0, 0, 0, 0};
#if __clang_major__ >= 23
    v8i z8 = {0, 0, 0, 0, 0, 0, 0, 0};
    __builtin_amdgcn_tensor_load_to_lds(g0, g1, z4, z4, z8, 0);
#else
    __builtin_amdgcn_tensor_load_to_lds(g0, g1, z4, z4, 0);
#endif
}

// One workgroup (8 waves, 256 threads) per (batch, 16-row q-tile).
__launch_bounds__(256, 1)
__global__ void sdpa_kernel(const float* __restrict__ Q,
                            const float* __restrict__ Kt,
                            const float* __restrict__ Vt,
                            const int*   __restrict__ Mask,
                            float* __restrict__ Out,
                            float* __restrict__ Wgt)
{
    __shared__ __align__(16) unsigned char arena[ARENA_SZ];
    float* s_strip = (float*)arena;
    float* red     = (float*)(arena + OFF_RED);
    float* rowmax  = (float*)(arena + OFF_RMAX);
    float* rowrcp  = (float*)(arena + OFF_RRCP);

    const int tid  = threadIdx.x;
    const int wave = tid >> 5;
    const int lane = tid & 31;
    const int m    = lane & 15;
    const int hi   = lane >> 4;

    const int b  = blockIdx.x >> 7;      // LQ/QTILE == 128
    const int qt = blockIdx.x & 127;
    const int q0 = qt * QTILE;
    const size_t qbase = (size_t)b * LQ + q0;

    // ---- Q^T B-fragments (32x16 f16), built once: B[k][n=lane&15] = Q[q0+n][kk+k], k in table order
    HV bQ[4];
    {
        const float* qb = Q + qbase * DH;
        #pragma unroll
        for (int kkI = 0; kkI < 4; ++kkI) {
            const int kk = kkI * 32 + hi * 8;
            #pragma unroll
            for (int j = 0; j < 8; ++j)
                bQ[kkI].h[j] = __builtin_amdgcn_cvt_pkrtz(qb[m*DH + kk + 2*j + (j >> 2) * 8],
                                                          qb[m*DH + kk + 2*j + 1 + (j >> 2) * 8]);
        }
    }

    // ---- Phase 1: S^T = K Q^T (scaled, unmasked) -> fp32 strip.  K tiles via TDM ----
    {
        float* kst0 = (float*)(arena + OFF_REGB + wave * (2 * KBUF_SZ));
        float* kst1 = (float*)(arena + OFF_REGB + wave * (2 * KBUF_SZ) + KBUF_SZ);
        const float* kbase = Kt + (size_t)b * LK * DH;
        tdm_load_2d(lds_off(kst0), kbase + (size_t)(wave * 256) * DH,
                    DH, QTILE, DH, QTILE, DH, /*pad: 4DW per 128DW*/ 6, 3);
        for (int t = 0; t < 16; ++t) {
            const int k0 = wave * 256 + t * 16;
            asm volatile("" ::: "memory");
            if (t + 1 < 16) {
                tdm_load_2d(lds_off((t & 1) ? kst0 : kst1),
                            kbase + (size_t)(k0 + 16) * DH,
                            DH, QTILE, DH, QTILE, DH, 6, 3);
                __builtin_amdgcn_s_wait_tensorcnt(1);
            } else {
                __builtin_amdgcn_s_wait_tensorcnt(0);
            }
            asm volatile("" ::: "memory");

            const float* ksf = (t & 1) ? kst1 : kst0;
            v8f acc = {};
            #pragma unroll
            for (int kkI = 0; kkI < 4; ++kkI) {
                // A (16x32 f16) from K tile row m: contiguous b128 LDS loads
                const float* krow = ksf + m * KSTR + kkI * 32 + hi * 8;
                const float4 a0 = *(const float4*)(krow);
                const float4 a1 = *(const float4*)(krow + 4);
                const float4 a2 = *(const float4*)(krow + 16);
                const float4 a3 = *(const float4*)(krow + 20);
                HV ak;
                ak.h[0] = __builtin_amdgcn_cvt_pkrtz(a0.x, a0.y);
                ak.h[1] = __builtin_amdgcn_cvt_pkrtz(a0.z, a0.w);
                ak.h[2] = __builtin_amdgcn_cvt_pkrtz(a1.x, a1.y);
                ak.h[3] = __builtin_amdgcn_cvt_pkrtz(a1.z, a1.w);
                ak.h[4] = __builtin_amdgcn_cvt_pkrtz(a2.x, a2.y);
                ak.h[5] = __builtin_amdgcn_cvt_pkrtz(a2.z, a2.w);
                ak.h[6] = __builtin_amdgcn_cvt_pkrtz(a3.x, a3.y);
                ak.h[7] = __builtin_amdgcn_cvt_pkrtz(a3.z, a3.w);
                acc = __builtin_amdgcn_wmma_f32_16x16x32_f16(
                    false, ak.v, false, bQ[kkI].v, (short)0, acc, false, false);
            }
            // S^T C-layout: q-row = lane&15, k-cols = k0 + 8*hi + v  -> contiguous b128 stores
            float4 s0, s1;
            s0.x = acc[0] * SCALE; s0.y = acc[1] * SCALE;
            s0.z = acc[2] * SCALE; s0.w = acc[3] * SCALE;
            s1.x = acc[4] * SCALE; s1.y = acc[5] * SCALE;
            s1.z = acc[6] * SCALE; s1.w = acc[7] * SCALE;
            float4* srow = (float4*)&s_strip[m * SSTRIDE + k0 + 8 * hi];
            srow[0] = s0;
            srow[1] = s1;
        }
    }
    __syncthreads();

    // ---- Phase 2: mask (coalesced int4) + row softmax, 16 threads/row ----
    {
        const int r = tid >> 4, sub = tid & 15;
        float4* srw = (float4*)&s_strip[r * SSTRIDE];
        const v4i* mrow = (const v4i*)(Mask + (qbase + r) * LK);
        float mx = -3.0e38f;
        for (int i = sub; i < LK / 4; i += 16) {
            float4 x = srw[i];
            const v4i mk = mrow[i];
            x.x = (mk[0] == 0) ? NEGINF : x.x;
            x.y = (mk[1] == 0) ? NEGINF : x.y;
            x.z = (mk[2] == 0) ? NEGINF : x.z;
            x.w = (mk[3] == 0) ? NEGINF : x.w;
            srw[i] = x;
            mx = fmaxf(fmaxf(fmaxf(mx, x.x), fmaxf(x.y, x.z)), x.w);
        }
        red[r * 16 + sub] = mx;
        __syncthreads();
        if (sub == 0) {
            float mm = red[r * 16];
            #pragma unroll
            for (int i = 1; i < 16; ++i) mm = fmaxf(mm, red[r * 16 + i]);
            rowmax[r] = mm;
        }
        __syncthreads();
        const float rm = rowmax[r];
        float sum = 0.f;
        for (int i = sub; i < LK / 4; i += 16) {
            float4 x = srw[i];
            x.x = __expf(x.x - rm); x.y = __expf(x.y - rm);
            x.z = __expf(x.z - rm); x.w = __expf(x.w - rm);
            srw[i] = x;
            sum += (x.x + x.y) + (x.z + x.w);
        }
        red[r * 16 + sub] = sum;
        __syncthreads();
        if (sub == 0) {
            float ss = 0.f;
            #pragma unroll
            for (int i = 0; i < 16; ++i) ss += red[r * 16 + i];
            rowrcp[r] = 1.0f / ss;
        }
        __syncthreads();
    }

    // ---- Phase 3: normalize; float4 weights store + packed-f16 P-strip ----
    {
        float4* wrow = (float4*)(Wgt + qbase * LK);
        _Float16* pst = (_Float16*)(arena + OFF_REGB);   // K stages are dead now
        for (int idx = tid; idx < QTILE * LK / 4; idx += 256) {
            const int r = idx >> 9, c4 = idx & 511;      // LK/4 == 512
            float4 p = ((float4*)&s_strip[r * SSTRIDE])[c4];
            const float rc = rowrcp[r];
            p.x *= rc; p.y *= rc; p.z *= rc; p.w *= rc;
            wrow[r * (LK / 4) + c4] = p;
            PK pk;
            pk.h[0] = __builtin_amdgcn_cvt_pkrtz(p.x, p.y);
            pk.h[1] = __builtin_amdgcn_cvt_pkrtz(p.z, p.w);
            *(v2u*)(pst + r * PSTRIDE + c4 * 4) = pk.u;
        }
    }
    __syncthreads();

    // ---- Phase 4: O = P V.  f16 A-fragments from P-strip; V tiles via TDM ----
    {
        const int n0 = wave * 16;
        const _Float16* pst = (const _Float16*)(arena + OFF_REGB);
        float* vst0 = (float*)(arena + OFF_VST + wave * (2 * VBUF_SZ));
        float* vst1 = (float*)(arena + OFF_VST + wave * (2 * VBUF_SZ) + VBUF_SZ);
        const float* vbase = Vt + (size_t)b * LK * DH + n0;
        tdm_load_2d(lds_off(vst0), vbase, 16, 32, 16, 32, DH, /*pad: 4DW per 16DW*/ 3, 3);
        v8f o = {};
        for (int s = 0; s < LK / 32; ++s) {
            asm volatile("" ::: "memory");
            if (s + 1 < LK / 32) {
                tdm_load_2d(lds_off((s & 1) ? vst0 : vst1),
                            vbase + (size_t)(s * 32 + 32) * DH, 16, 32, 16, 32, DH, 3, 3);
                __builtin_amdgcn_s_wait_tensorcnt(1);
            } else {
                __builtin_amdgcn_s_wait_tensorcnt(0);
            }
            asm volatile("" ::: "memory");
            const float* vt = (s & 1) ? vst1 : vst0;

            // A fragment: two raw b128 loads from the f16 P-strip (no conversion)
            HV ap;
            const _Float16* prow = pst + m * PSTRIDE + s * 32 + hi * 8;
            ap.q[0] = *(const v4u*)prow;
            ap.q[1] = *(const v4u*)(prow + 16);
            // B fragment: n = lane&15; padded V-tile column, k in table order
            const float* vcol = vt + m;
            HV bv;
            #pragma unroll
            for (int j = 0; j < 8; ++j) {
                const int kb = hi * 8 + 2 * j + (j >> 2) * 8;
                bv.h[j] = __builtin_amdgcn_cvt_pkrtz(vcol[kb * VSTR], vcol[(kb + 1) * VSTR]);
            }
            o = __builtin_amdgcn_wmma_f32_16x16x32_f16(
                false, ap.v, false, bv.v, (short)0, o, false, false);
        }
        #pragma unroll
        for (int v = 0; v < 8; ++v)
            Out[(qbase + v + 8 * hi) * DH + n0 + m] = o[v];
    }
}

extern "C" void kernel_launch(void* const* d_in, const int* in_sizes, int n_in,
                              void* d_out, int out_size, void* d_ws, size_t ws_size,
                              hipStream_t stream) {
    const float* Q  = (const float*)d_in[0];
    const float* Kt = (const float*)d_in[1];
    const float* Vt = (const float*)d_in[2];
    const int*   M  = (const int*)d_in[3];
    float* Out = (float*)d_out;
    float* Wgt = Out + (size_t)B_ * LQ * DH;   // outputs concatenated: O, then P

    dim3 grid(B_ * (LQ / QTILE));              // 1024 workgroups
    sdpa_kernel<<<grid, 256, 0, stream>>>(Q, Kt, Vt, M, Out, Wgt);
}
